// MultiHeadAttention_64029372449073
// MI455X (gfx1250) — matrix operand order repaired and checked by reading, not compiled, not run on hardware
//
#include <hip/hip_runtime.h>

// ---------------------------------------------------------------------------
// MI455X (gfx1250) multi-head attention:
//   1) qkv_proj_kernel : fp32 GEMM (x @ W^T + b) -> bf16, WMMA f32_16x16x32_bf16
//                        Q,K written [B,H,S,D]; V written transposed [B,H,D,S]
//   2) attention_kernel: flash-attention per (b,h,64-q-rows), WMMA QK^T and PV,
//                        K/Vt tiles staged via Tensor Data Mover (TENSORcnt)
//   3) out_proj_kernel : bf16 GEMM (attn @ Wo^T + bo) -> fp32 d_out, A tile TDM
// ---------------------------------------------------------------------------

#define EMBED 1024
#define HEADS 16
#define HDIM  64
#define BATCH 4
#define SEQ   2048
#define MTOT  (BATCH * SEQ)   // 8192

typedef __attribute__((ext_vector_type(16))) __bf16 v16bf;
typedef __attribute__((ext_vector_type(8)))  __bf16 v8bf;
typedef __attribute__((ext_vector_type(8)))  float  v8f;
typedef __attribute__((ext_vector_type(4)))  unsigned v4u;
typedef __attribute__((ext_vector_type(8)))  unsigned v8u;

union Frag16 { v16bf v; v8bf h[2]; };

__device__ __forceinline__ __bf16 f2bf(float f) { return (__bf16)f; }

__device__ __forceinline__ v8f wmma_bf16(v16bf a, v16bf b, v8f c) {
    // D(f32 16x16) = A(bf16 16x32) * B(bf16 32x16) + C
    return __builtin_amdgcn_wmma_f32_16x16x32_bf16(
        /*neg_a=*/false, a, /*neg_b=*/false, b,
        /*c_mod=*/(short)0, c, /*reuse_a=*/false, /*reuse_b=*/false);
}

// ---------------------------------------------------------------------------
// Tensor Data Mover: 2-D tile load, bf16 (data_size=2B), row padding in LDS.
// tile_w elements per row, tile_h rows, stride_elems between rows in memory.
// pad_i: pad interval code (3 => every 16 DWORDs, 4 => every 32 DWORDs)
// pad_a: pad amount   code (3 => 4 DWORDs = 16B = 8 halves)
// Issue from ONE wave only; complete with s_wait_tensorcnt.
// ---------------------------------------------------------------------------
__device__ __forceinline__ void tdm_load_2d_bf16(unsigned lds_off, const void* gptr,
                                                 unsigned tile_w, unsigned tile_h,
                                                 unsigned stride_elems,
                                                 unsigned pad_i, unsigned pad_a)
{
    unsigned long long ga = (unsigned long long)gptr;
    v4u g0;
    g0.x = 1u;                                        // count=1 (valid D#)
    g0.y = lds_off;                                   // lds_addr (bytes)
    g0.z = (unsigned)ga;                              // global_addr[31:0]
    g0.w = (unsigned)((ga >> 32) & 0x01FFFFFFu)       // global_addr[56:32]
         | 0x80000000u;                               // type=2 ("image")
    v8u g1;
    g1.s0 = (1u << 16)                                // data_size = 2 bytes
          | (1u << 20)                                // pad_enable
          | (pad_i << 22) | (pad_a << 25);            // LDS row padding
    g1.s1 = (tile_w & 0xFFFFu) << 16;                 // tensor_dim0[15:0]
    g1.s2 = (tile_h & 0xFFFFu) << 16;                 // dim0 hi=0 | tensor_dim1 lo
    g1.s3 = (tile_w & 0xFFFFu) << 16;                 // dim1 hi=0 | tile_dim0
    g1.s4 = (tile_h & 0xFFFFu);                       // tile_dim1 | tile_dim2=0
    g1.s5 = stride_elems;                             // tensor_dim0_stride[31:0]
    g1.s6 = 0u;                                       // stride hi | dim1_stride lo
    g1.s7 = 0u;
    asm volatile("tensor_load_to_lds %0, %1" :: "s"(g0), "s"(g1) : "memory");
}

__device__ __forceinline__ unsigned lds_offset_of(const void* p) {
    // LDS aperture maps the low 32 bits of the generic address to the LDS offset
    return (unsigned)(unsigned long long)p;
}

// ---------------------------------------------------------------------------
// Kernel 1: fused QKV projections.  C[m,n] = sum_k A[m,k]*W[n,k] + bias[n]
// Block tile 128(M) x 64(N), K-step 32, 8 waves each owning a 32x32 subtile.
// blockIdx.z: 0=Q, 1=K, 2=V(transposed output).
// ---------------------------------------------------------------------------
__global__ __launch_bounds__(256)
void qkv_proj_kernel(const float* __restrict__ xq, const float* __restrict__ xk,
                     const float* __restrict__ xv,
                     const float* __restrict__ Wq, const float* __restrict__ Wk,
                     const float* __restrict__ Wv,
                     const float* __restrict__ bq, const float* __restrict__ bk,
                     const float* __restrict__ bv,
                     __bf16* __restrict__ Qb, __bf16* __restrict__ Kb,
                     __bf16* __restrict__ Vtb)
{
    constexpr int BM = 128, BN = 64, BK = 32, LA = 40, LW = 40;
    __shared__ __bf16 As[BM][LA];   // activations tile, bf16
    __shared__ __bf16 Ws[BN][LW];   // weight tile, row n = output feature

    const int mode = blockIdx.z;
    const float* A    = (mode == 0) ? xq : (mode == 1) ? xk : xv;
    const float* W    = (mode == 0) ? Wq : (mode == 1) ? Wk : Wv;
    const float* bias = (mode == 0) ? bq : (mode == 1) ? bk : bv;
    __bf16*      out  = (mode == 0) ? Qb : (mode == 1) ? Kb : Vtb;

    const int mbase = blockIdx.y * BM;
    const int nbase = blockIdx.x * BN;
    const int tid  = threadIdx.x;
    const int lane = tid & 31;
    const int wave = tid >> 5;
    const int wm = wave >> 1;       // 0..3  (M subtile)
    const int wn = wave & 1;        // 0..1  (N subtile)

    v8f acc[2][2] = {};

    for (int k0 = 0; k0 < EMBED; k0 += BK) {
        // stage A: 128x32 fp32 -> bf16 (4 float4 per thread)
        #pragma unroll
        for (int i = 0; i < 4; ++i) {
            int vid = i * 256 + tid;
            int row = vid >> 3;
            int col = (vid & 7) * 4;
            float4 a4 = *(const float4*)(A + (size_t)(mbase + row) * EMBED + k0 + col);
            union { __bf16 b[4]; uint2 u; } pk;
            pk.b[0] = f2bf(a4.x); pk.b[1] = f2bf(a4.y);
            pk.b[2] = f2bf(a4.z); pk.b[3] = f2bf(a4.w);
            *(uint2*)&As[row][col] = pk.u;
        }
        // stage W: 64x32 fp32 -> bf16 (2 float4 per thread)
        #pragma unroll
        for (int i = 0; i < 2; ++i) {
            int vid = i * 256 + tid;
            int row = vid >> 3;
            int col = (vid & 7) * 4;
            float4 a4 = *(const float4*)(W + (size_t)(nbase + row) * EMBED + k0 + col);
            union { __bf16 b[4]; uint2 u; } pk;
            pk.b[0] = f2bf(a4.x); pk.b[1] = f2bf(a4.y);
            pk.b[2] = f2bf(a4.z); pk.b[3] = f2bf(a4.w);
            *(uint2*)&Ws[row][col] = pk.u;
        }
        __syncthreads();

        Frag16 af[2], bf[2];
        #pragma unroll
        for (int mi = 0; mi < 2; ++mi) {
            int r = wm * 32 + mi * 16 + (lane & 15);
            int o = (lane >> 4) * 8;
            af[mi].h[0] = *(const v8bf*)&As[r][o];        // K = o..o+7
            af[mi].h[1] = *(const v8bf*)&As[r][16 + o];   // K = 16+o..16+o+7
        }
        #pragma unroll
        for (int ni = 0; ni < 2; ++ni) {
            int n  = wn * 32 + ni * 16 + (lane & 15);
            int ko = (lane >> 4) * 8;
            bf[ni].h[0] = *(const v8bf*)&Ws[n][ko];       // K = ko..ko+7
            bf[ni].h[1] = *(const v8bf*)&Ws[n][ko + 16];  // K = 16+ko..16+ko+7
        }
        #pragma unroll
        for (int mi = 0; mi < 2; ++mi)
            #pragma unroll
            for (int ni = 0; ni < 2; ++ni)
                acc[mi][ni] = wmma_bf16(af[mi].v, bf[ni].v, acc[mi][ni]);
        __syncthreads();
    }

    // epilogue: scatter into head-major bf16 layouts
    #pragma unroll
    for (int ni = 0; ni < 2; ++ni) {
        int n = nbase + wn * 32 + ni * 16 + (lane & 15);
        float bv_ = bias[n];
        int h = n >> 6;       // head
        int d = n & 63;       // dim inside head
        #pragma unroll
        for (int mi = 0; mi < 2; ++mi) {
            #pragma unroll
            for (int g = 0; g < 8; ++g) {
                int m = mbase + wm * 32 + mi * 16 + (lane >> 4) * 8 + g;
                int b = m >> 11;          // batch (S=2048)
                int s = m & 2047;
                float val = acc[mi][ni][g] + bv_;
                size_t idx;
                if (mode == 2)   // V transposed: [B,H,D,S]
                    idx = ((((size_t)b * HEADS + h) * HDIM + d) * SEQ + s);
                else             // Q,K natural: [B,H,S,D]
                    idx = ((((size_t)b * HEADS + h) * SEQ + s) * HDIM + d);
                out[idx] = f2bf(val);
            }
        }
    }
}

// ---------------------------------------------------------------------------
// Kernel 2: flash attention.  One block = one (b,h) x 64 query rows.
// 4 waves, each wave owns 16 query rows.  Key tile = 64, staged by TDM.
// ---------------------------------------------------------------------------
__global__ __launch_bounds__(128)
void attention_kernel(const __bf16* __restrict__ Qb, const __bf16* __restrict__ Kb,
                      const __bf16* __restrict__ Vtb, const int* __restrict__ mask,
                      __bf16* __restrict__ Ab)
{
    constexpr int LT = 72;               // LDS row stride (halves): 32+4 DWORDs
    __shared__ __bf16 Ks[64][LT];        // K tile   [key][d]
    __shared__ __bf16 Vs[64][LT];        // Vt tile  [d][key]
    __shared__ __bf16 Ps[4][16][LT];     // per-wave P scratch [qrow][key]

    const int tid  = threadIdx.x;
    const int lane = tid & 31;
    const int wave = tid >> 5;
    const int qbase = blockIdx.x * 64;
    const int bh = blockIdx.y;
    const int b = bh >> 4;
    const int h = bh & 15;
    const float scale = 0.125f;          // 1/sqrt(64)

    const size_t headQK = (((size_t)b * HEADS + h) * SEQ) * HDIM; // Q,K base
    const size_t headVT = (((size_t)b * HEADS + h) * HDIM) * SEQ; // Vt base

    // resident Q fragments (rows qbase+wave*16 .. +15, D=64 -> 2 K-slices)
    Frag16 qa[2];
    {
        int r = qbase + wave * 16 + (lane & 15);
        int o = (lane >> 4) * 8;
        #pragma unroll
        for (int ks = 0; ks < 2; ++ks) {
            const __bf16* p = Qb + headQK + (size_t)r * HDIM + ks * 32;
            qa[ks].h[0] = *(const v8bf*)(p + o);
            qa[ks].h[1] = *(const v8bf*)(p + 16 + o);
        }
    }

    const unsigned ldsK = lds_offset_of(&Ks[0][0]);
    const unsigned ldsV = lds_offset_of(&Vs[0][0]);

    float mrow[8], lrow[8];
    #pragma unroll
    for (int g = 0; g < 8; ++g) { mrow[g] = -3.0e38f; lrow[g] = 0.0f; }
    v8f oacc[4] = {};

    for (int kt = 0; kt < SEQ / 64; ++kt) {
        const int sk = kt * 64;
        // TDM: stage K tile (64 keys x 64 d, row stride 64) and
        //      Vt tile (64 d x 64 keys, row stride S). Wave 0 issues the DMA;
        //      row pad: every 32 DWORDs insert 4 DWORDs -> LDS stride 72 halves.
        if (wave == 0) {
            tdm_load_2d_bf16(ldsK, Kb + headQK + (size_t)sk * HDIM,
                             /*w=*/64, /*h=*/64, /*stride=*/HDIM, /*pi=*/4, /*pa=*/3);
            tdm_load_2d_bf16(ldsV, Vtb + headVT + sk,
                             /*w=*/64, /*h=*/64, /*stride=*/SEQ, /*pi=*/4, /*pa=*/3);
            __builtin_amdgcn_s_wait_tensorcnt(0);
        }
        if (kt + 1 < SEQ / 64) {  // hint next tiles (global_prefetch_b8)
            __builtin_prefetch(Kb + headQK + (size_t)(sk + 64) * HDIM, 0, 1);
            __builtin_prefetch(Vtb + headVT + sk + 64, 0, 1);
        }
        __syncthreads();

        // energy E = Q * K^T  (4 fragments of 16 keys, 2 d-slices each)
        v8f E[4] = {};
        #pragma unroll
        for (int f = 0; f < 4; ++f) {
            #pragma unroll
            for (int ks = 0; ks < 2; ++ks) {
                Frag16 bf;
                int n  = f * 16 + (lane & 15);       // key within tile
                int ko = ks * 32 + (lane >> 4) * 8;  // d offset
                bf.h[0] = *(const v8bf*)&Ks[n][ko];
                bf.h[1] = *(const v8bf*)&Ks[n][ko + 16];
                E[f] = wmma_bf16(qa[ks].v, bf.v, E[f]);
            }
        }
        // mask + scale (column is lane-uniform per fragment)
        #pragma unroll
        for (int f = 0; f < 4; ++f) {
            int key = sk + f * 16 + (lane & 15);
            int mv = mask[(size_t)b * SEQ + key];
            #pragma unroll
            for (int g = 0; g < 8; ++g)
                E[f][g] = mv ? E[f][g] * scale : -1.25e19f;
        }
        // streaming softmax: per-row stats via 16-lane butterfly
        #pragma unroll
        for (int g = 0; g < 8; ++g) {
            float tm = fmaxf(fmaxf(E[0][g], E[1][g]), fmaxf(E[2][g], E[3][g]));
            tm = fmaxf(tm, __shfl_xor(tm, 1, 32));
            tm = fmaxf(tm, __shfl_xor(tm, 2, 32));
            tm = fmaxf(tm, __shfl_xor(tm, 4, 32));
            tm = fmaxf(tm, __shfl_xor(tm, 8, 32));
            float newm  = fmaxf(mrow[g], tm);
            float alpha = __expf(mrow[g] - newm);
            mrow[g] = newm;
            float ssum = 0.0f;
            #pragma unroll
            for (int f = 0; f < 4; ++f) {
                float p = __expf(E[f][g] - newm);
                E[f][g] = p;
                ssum += p;
            }
            ssum += __shfl_xor(ssum, 1, 32);
            ssum += __shfl_xor(ssum, 2, 32);
            ssum += __shfl_xor(ssum, 4, 32);
            ssum += __shfl_xor(ssum, 8, 32);
            lrow[g] = lrow[g] * alpha + ssum;
            #pragma unroll
            for (int nf = 0; nf < 4; ++nf)
                oacc[nf][g] *= alpha;
            // spill P (C-layout) to LDS so it can be re-read in A-layout
            int prow = (lane >> 4) * 8 + g;
            #pragma unroll
            for (int f = 0; f < 4; ++f)
                Ps[wave][prow][f * 16 + (lane & 15)] = f2bf(E[f][g]);
        }
        __syncthreads();

        // O += P * V   (P: 16x64 bf16, Vt rows give B operand contiguously)
        #pragma unroll
        for (int ks = 0; ks < 2; ++ks) {
            Frag16 pa;
            int r = lane & 15;
            int o = (lane >> 4) * 8;
            pa.h[0] = *(const v8bf*)&Ps[wave][r][ks * 32 + o];
            pa.h[1] = *(const v8bf*)&Ps[wave][r][ks * 32 + 16 + o];
            #pragma unroll
            for (int nf = 0; nf < 4; ++nf) {
                Frag16 bf;
                int dcol = nf * 16 + (lane & 15);         // output dim
                int ko   = ks * 32 + (lane >> 4) * 8;     // key offset
                bf.h[0] = *(const v8bf*)&Vs[dcol][ko];
                bf.h[1] = *(const v8bf*)&Vs[dcol][ko + 16];
                oacc[nf] = wmma_bf16(pa.v, bf.v, oacc[nf]);
            }
        }
        __syncthreads();
    }

    // normalize and store concat-head bf16 [B,S,EMBED]
    #pragma unroll
    for (int nf = 0; nf < 4; ++nf) {
        int d = nf * 16 + (lane & 15);
        #pragma unroll
        for (int g = 0; g < 8; ++g) {
            int s = qbase + wave * 16 + (lane >> 4) * 8 + g;
            float val = oacc[nf][g] / lrow[g];
            Ab[((size_t)b * SEQ + s) * EMBED + h * HDIM + d] = f2bf(val);
        }
    }
}

// ---------------------------------------------------------------------------
// Kernel 3: output projection.  out[m,n] = sum_k Ab[m,k]*Wo[n,k] + bo[n] (fp32)
// bf16 A tile staged by TDM; fp32 W tile converted by VALU.
// ---------------------------------------------------------------------------
__global__ __launch_bounds__(256)
void out_proj_kernel(const __bf16* __restrict__ A, const float* __restrict__ W,
                     const float* __restrict__ bias, float* __restrict__ out)
{
    constexpr int BM = 128, BN = 64, BK = 32, LA = 40, LW = 40;
    __shared__ __bf16 As[BM][LA];    // row = 16 DWORDs data + 4 DWORDs pad
    __shared__ __bf16 Ws[BN][LW];

    const int mbase = blockIdx.y * BM;
    const int nbase = blockIdx.x * BN;
    const int tid  = threadIdx.x;
    const int lane = tid & 31;
    const int wave = tid >> 5;
    const int wm = wave >> 1;
    const int wn = wave & 1;

    const unsigned ldsA = lds_offset_of(&As[0][0]);

    v8f acc[2][2] = {};

    for (int k0 = 0; k0 < EMBED; k0 += BK) {
        // TDM: stage A tile 128 rows x 32 halves, row stride EMBED halves;
        // pad every 16 DWORDs by 4 DWORDs -> LDS stride 40 halves.
        if (wave == 0) {
            tdm_load_2d_bf16(ldsA, A + (size_t)mbase * EMBED + k0,
                             /*w=*/BK, /*h=*/BM, /*stride=*/EMBED, /*pi=*/3, /*pa=*/3);
        }
        // stage W: 64x32 fp32 -> bf16
        #pragma unroll
        for (int i = 0; i < 2; ++i) {
            int vid = i * 256 + tid;
            int row = vid >> 3;
            int col = (vid & 7) * 4;
            float4 a4 = *(const float4*)(W + (size_t)(nbase + row) * EMBED + k0 + col);
            union { __bf16 b[4]; uint2 u; } pk;
            pk.b[0] = f2bf(a4.x); pk.b[1] = f2bf(a4.y);
            pk.b[2] = f2bf(a4.z); pk.b[3] = f2bf(a4.w);
            *(uint2*)&Ws[row][col] = pk.u;
        }
        if (wave == 0) __builtin_amdgcn_s_wait_tensorcnt(0);
        __syncthreads();

        Frag16 af[2], bf[2];
        #pragma unroll
        for (int mi = 0; mi < 2; ++mi) {
            int r = wm * 32 + mi * 16 + (lane & 15);
            int o = (lane >> 4) * 8;
            af[mi].h[0] = *(const v8bf*)&As[r][o];
            af[mi].h[1] = *(const v8bf*)&As[r][16 + o];
        }
        #pragma unroll
        for (int ni = 0; ni < 2; ++ni) {
            int n  = wn * 32 + ni * 16 + (lane & 15);
            int ko = (lane >> 4) * 8;
            bf[ni].h[0] = *(const v8bf*)&Ws[n][ko];
            bf[ni].h[1] = *(const v8bf*)&Ws[n][ko + 16];
        }
        #pragma unroll
        for (int mi = 0; mi < 2; ++mi)
            #pragma unroll
            for (int ni = 0; ni < 2; ++ni)
                acc[mi][ni] = wmma_bf16(af[mi].v, bf[ni].v, acc[mi][ni]);
        __syncthreads();
    }

    #pragma unroll
    for (int ni = 0; ni < 2; ++ni) {
        int n = nbase + wn * 32 + ni * 16 + (lane & 15);
        float bv_ = bias[n];
        #pragma unroll
        for (int mi = 0; mi < 2; ++mi) {
            #pragma unroll
            for (int g = 0; g < 8; ++g) {
                int m = mbase + wm * 32 + mi * 16 + (lane >> 4) * 8 + g;
                out[(size_t)m * EMBED + n] = acc[mi][ni][g] + bv_;
            }
        }
    }
}

// ---------------------------------------------------------------------------
extern "C" void kernel_launch(void* const* d_in, const int* in_sizes, int n_in,
                              void* d_out, int out_size, void* d_ws, size_t ws_size,
                              hipStream_t stream)
{
    const float* query  = (const float*)d_in[0];
    const float* keys   = (const float*)d_in[1];
    const float* values = (const float*)d_in[2];
    const int*   mask   = (const int*)d_in[3];
    const float* Wq = (const float*)d_in[4];  const float* bq = (const float*)d_in[5];
    const float* Wk = (const float*)d_in[6];  const float* bk = (const float*)d_in[7];
    const float* Wv = (const float*)d_in[8];  const float* bv = (const float*)d_in[9];
    const float* Wo = (const float*)d_in[10]; const float* bo = (const float*)d_in[11];

    // workspace layout: 4 x 16 MiB bf16 buffers (Q, K, Vt, attn_out)
    char* w = (char*)d_ws;
    __bf16* Qb  = (__bf16*)(w + 0 * (1ull << 24));
    __bf16* Kb  = (__bf16*)(w + 1 * (1ull << 24));
    __bf16* Vtb = (__bf16*)(w + 2 * (1ull << 24));
    __bf16* Ab  = (__bf16*)(w + 3 * (1ull << 24));

    // 1) QKV projections: grid = (N/64, M/128, 3 modes)
    qkv_proj_kernel<<<dim3(EMBED / 64, MTOT / 128, 3), 256, 0, stream>>>(
        query, keys, values, Wq, Wk, Wv, bq, bk, bv, Qb, Kb, Vtb);

    // 2) flash attention: grid = (S/64, B*H)
    attention_kernel<<<dim3(SEQ / 64, BATCH * HEADS), 128, 0, stream>>>(
        Qb, Kb, Vtb, mask, Ab);

    // 3) output projection into d_out (fp32)
    out_proj_kernel<<<dim3(EMBED / 64, MTOT / 128), 256, 0, stream>>>(
        Ab, Wo, bo, (float*)d_out);
}
